// KernelPointAggregation_5205500363353
// MI455X (gfx1250) — matrix operand order repaired and checked by reading, not compiled, not run on hardware
//
#include <hip/hip_runtime.h>

typedef __attribute__((ext_vector_type(16))) _Float16 v16h;
typedef __attribute__((ext_vector_type(8)))  _Float16 v8h;
typedef __attribute__((ext_vector_type(8)))  float    v8f;

#define NEI 32
#define KK  8
#define DD  64

// ---------------------------------------------------------------------------
// Prep kernel (runs once per launch): W fp32 -> f16, kmet = kernels*metric,
// expsc = exp(scales). All loop-invariant across the 12000 point-blocks.
// ---------------------------------------------------------------------------
__global__ __launch_bounds__(256) void kpagg_prep(
    const float* __restrict__ W,       // (8, 64, 64)
    const float* __restrict__ kp,      // (8, 64)
    const float* __restrict__ scales,  // (8,)
    _Float16* __restrict__ Wh,         // (8*64*64) f16
    float* __restrict__ kmet,          // (8*64)
    float* __restrict__ expsc)         // (8)
{
    const int gid = blockIdx.x * 256 + threadIdx.x;   // 16 blocks -> 4096 threads

    // convert 8 W elements each (4096 * 8 = 32768)
    #pragma unroll
    for (int i = 0; i < 8; ++i) {
        int idx = gid * 8 + i;
        Wh[idx] = (_Float16)W[idx];
    }

    if (gid < KK * DD) {
        int k = gid >> 6, d = gid & 63;
        float s2 = 0.0f;
        for (int dd = 1; dd < DD; ++dd) { float v = kp[k * DD + dd]; s2 += v * v; }
        float nrm = sqrtf(fmaxf(s2, 1e-8f));
        float sns = sinhf(nrm) / nrm;
        kmet[gid] = (d == 0) ? coshf(nrm) : -sns * kp[gid];
    }
    if (gid < KK) expsc[gid] = expf(scales[gid]);
}

// ---------------------------------------------------------------------------
// Main kernel: one 256-thread block (8 wave32) per point n; wave w owns k=w.
// ---------------------------------------------------------------------------
__global__ __launch_bounds__(256) void kpagg_kernel(
    const float* __restrict__ x,          // (N, 64)
    const long long* __restrict__ nei,    // (N, 32) int64
    const int* __restrict__ nei_mask,     // (N, 32) int32
    const _Float16* __restrict__ Wh,      // (8, 64, 64) f16 (prepped)
    const float* __restrict__ kmet,       // (8, 64)       (prepped)
    const float* __restrict__ expsc,      // (8,)          (prepped)
    const float* __restrict__ bias,       // (8, 64)
    float* __restrict__ out)              // (N, 64)
{
    __shared__ __align__(16) float    Asf[NEI * DD];   // gathered rows, fp32
    __shared__ __align__(16) _Float16 Ash[NEI * DD];   // gathered rows, f16 (WMMA A)
    __shared__ float diswk[KK * NEI];                  // dis, then softmax weights
    __shared__ __align__(16) float agg[NEI * DD];      // sum_k wk * x_tr
    __shared__ float wod[NEI];                         // (mask+1e-4)/lorentz_den
    __shared__ float mvec[DD];
    __shared__ float invden;

    const int n    = blockIdx.x;
    const int tid  = threadIdx.x;
    const int lane = tid & 31;
    const int wave = tid >> 5;       // == kernel index k for the GEMM phase
    const int g    = lane >> 4;      // half-wave group (0/1)
    const int nl   = lane & 15;      // N/M position within a 16-wide tile

    // ---------------- Phase 1: gather neighbor rows
    {
        const int row = tid >> 3;          // 0..31
        const int seg = tid & 7;           // 8 floats each
        const long long gi = nei[(long long)n * NEI + row];
        const float4* src = (const float4*)(x + gi * DD + seg * 8);
        float4 a0 = src[0];
        float4 a1 = src[1];
        float* df = &Asf[row * DD + seg * 8];
        ((float4*)df)[0] = a0;
        ((float4*)df)[1] = a1;
        _Float16* dh = &Ash[row * DD + seg * 8];
        dh[0] = (_Float16)a0.x; dh[1] = (_Float16)a0.y;
        dh[2] = (_Float16)a0.z; dh[3] = (_Float16)a0.w;
        dh[4] = (_Float16)a1.x; dh[5] = (_Float16)a1.y;
        dh[6] = (_Float16)a1.z; dh[7] = (_Float16)a1.w;

        #pragma unroll
        for (int i = 0; i < 8; ++i) agg[tid * 8 + i] = 0.0f;
    }
    __syncthreads();

    // ---------------- Phase 2: dis[k][m] = arccosh(clip(<x_row, kmet[k]>))
    {
        const int k = tid >> 5, m = tid & 31;
        const float* xr = &Asf[m * DD];
        const float* km = &kmet[k * DD];
        float ni = 0.0f;
        for (int d = 0; d < DD; ++d) ni += km[d] * xr[d];
        float z = fmaxf(ni, 1.0f + 1e-7f);
        diswk[k * NEI + m] = acoshf(z);
    }
    __syncthreads();

    // ---------------- Phase 3: softmax over k (per neighbor m)
    if (tid < NEI) {
        float e[KK]; float mx = -1e30f;
        #pragma unroll
        for (int k = 0; k < KK; ++k) { e[k] = -diswk[k * NEI + tid]; mx = fmaxf(mx, e[k]); }
        float s = 0.0f;
        #pragma unroll
        for (int k = 0; k < KK; ++k) { e[k] = expf(e[k] - mx); s += e[k]; }
        float inv = 1.0f / s;
        #pragma unroll
        for (int k = 0; k < KK; ++k) diswk[k * NEI + tid] = e[k] * inv;
    }
    __syncthreads();

    // ---------------- Phase 4: per-wave GEMM  y[m,o] = A(32x64) * W[k]^T + b[k]
    {
        const int k = wave;

        // B fragments straight from preconverted f16 W: fragment half h maps to
        // K = s*32 + 16*g + h, which is 16 contiguous d's -> one 32B v16h load.
        v16h Bf[2][4];
        #pragma unroll
        for (int s = 0; s < 2; ++s) {
            #pragma unroll
            for (int c = 0; c < 4; ++c) {
                Bf[s][c] = *(const v16h*)(Wh + ((k * DD + (c * 16 + nl)) * DD + s * 32 + 16 * g));
            }
        }

        // Accumulators seeded with the bias (b[k][o] broadcast over rows).
        v8f acc[2][4];
        #pragma unroll
        for (int t = 0; t < 2; ++t) {
            #pragma unroll
            for (int c = 0; c < 4; ++c) {
                float bv = bias[k * DD + c * 16 + nl];
                v8f a;
                #pragma unroll
                for (int r = 0; r < 8; ++r) a[r] = bv;
                acc[t][c] = a;
            }
        }

        // A fragments from LDS (k order chosen here, same as B: per lane one contiguous
        // 16-half run: d = s*32+16g..+15, matching B).
        #pragma unroll
        for (int t = 0; t < 2; ++t) {
            #pragma unroll
            for (int s = 0; s < 2; ++s) {
                const int row = t * 16 + nl;
                union { v16h v; v8h h[2]; } au;
                au.h[0] = *(const v8h*)(Ash + row * DD + s * 32 + 16 * g);
                au.h[1] = *(const v8h*)(Ash + row * DD + s * 32 + 16 * g + 8);
                #pragma unroll
                for (int c = 0; c < 4; ++c) {
                    acc[t][c] = __builtin_amdgcn_wmma_f32_16x16x32_f16(
                        false, au.v, false, Bf[s][c],
                        (short)0, acc[t][c], false, false);
                }
            }
        }

        // Epilogue in registers: C/D layout -> lane (g,nl), VGPR r holds
        // (row m = t*16 + r + 8g, col o = c*16 + nl).
        const float es = expsc[k];
        #pragma unroll
        for (int t = 0; t < 2; ++t) {
            #pragma unroll
            for (int r = 0; r < 8; ++r) {
                float p = 0.0f;
                #pragma unroll
                for (int c = 0; c < 4; ++c) {
                    float v  = acc[t][c][r];
                    float vv = v * v;
                    if (c == 0) vv = (nl == 0) ? 0.0f : vv;   // exclude o==0
                    p += vv;
                }
                p += __shfl_xor(p, 1, 32);
                p += __shfl_xor(p, 2, 32);
                p += __shfl_xor(p, 4, 32);
                p += __shfl_xor(p, 8, 32);
                // broadcast y[...,0] (held at lane g*16, c=0) to the half-wave
                float y0   = __shfl(acc[t][0][r], (lane & 16), 32);
                float time = es / (1.0f + expf(-y0)) + 1.0001f;
                float sc   = (time * time - 1.0f) / fmaxf(p, 1e-8f);
                float ssc  = sqrtf(sc);
                const int m = t * 16 + r + 8 * g;
                float wkm  = diswk[k * NEI + m];
                #pragma unroll
                for (int c = 0; c < 4; ++c) {
                    int   o   = c * 16 + nl;
                    float v   = acc[t][c][r];
                    float xtr = (o == 0) ? time : v * ssc;
                    atomicAdd(&agg[m * DD + o], wkm * xtr);
                }
            }
        }
    }
    __syncthreads();

    // ---------------- Phase 5: per-row Lorentz normalization + neighbor weight
    if (tid < NEI) {
        const float* a = &agg[tid * DD];
        float ln = -a[0] * a[0];
        for (int d = 1; d < DD; ++d) ln += a[d] * a[d];
        float den = sqrtf(fmaxf(fabsf(ln), 1e-8f));
        float wn  = (float)nei_mask[(long long)n * NEI + tid] + 1e-4f;
        wod[tid]  = wn / den;
    }
    __syncthreads();

    // ---------------- Phase 6: weighted midpoint over the 32 neighbors
    if (tid < DD) {
        float s = 0.0f;
        for (int m = 0; m < NEI; ++m) s += agg[m * DD + tid] * wod[m];
        mvec[tid] = s;
    }
    __syncthreads();

    // ---------------- Phase 7: final Lorentz normalization
    if (tid == 0) {
        float ln = -mvec[0] * mvec[0];
        for (int d = 1; d < DD; ++d) ln += mvec[d] * mvec[d];
        invden = 1.0f / sqrtf(fmaxf(fabsf(ln), 1e-8f));
    }
    __syncthreads();

    if (tid < DD) out[(long long)n * DD + tid] = mvec[tid] * invden;
}

extern "C" void kernel_launch(void* const* d_in, const int* in_sizes, int n_in,
                              void* d_out, int out_size, void* d_ws, size_t ws_size,
                              hipStream_t stream) {
    const float*     x     = (const float*)d_in[0];
    const long long* nei   = (const long long*)d_in[1];
    const int*       nmask = (const int*)d_in[2];
    const float*     kp    = (const float*)d_in[3];
    const float*     W     = (const float*)d_in[4];
    const float*     b     = (const float*)d_in[5];
    const float*     sc    = (const float*)d_in[6];
    float*           out   = (float*)d_out;

    // Workspace carve-out: Wh (64 KB f16) | kmet (2 KB) | expsc (32 B)
    _Float16* Wh    = (_Float16*)d_ws;
    float*    kmet  = (float*)((char*)d_ws + KK * DD * DD * sizeof(_Float16));
    float*    expsc = kmet + KK * DD;

    const int n_pts = in_sizes[0] / DD;   // 12000

    kpagg_prep<<<dim3(16), dim3(256), 0, stream>>>(W, kp, sc, Wh, kmet, expsc);
    kpagg_kernel<<<dim3(n_pts), dim3(256), 0, stream>>>(x, nei, nmask, Wh, kmet, expsc, b, out);
}
